// MyRAWLSTM_86234353369951
// MI455X (gfx1250) — matrix operand order repaired and checked by reading, not compiled, not run on hardware
//
#include <hip/hip_runtime.h>
#include <math.h>

// -------- problem constants --------
constexpr int Bb  = 64;    // batch
constexpr int Pp  = 12;    // encoder steps
constexpr int Qq  = 12;    // decoder steps
constexpr int Dd  = 64;    // hidden
constexpr int Nn  = 1024;  // cells (H*W)
constexpr int K3D = 192;   // 3*D
constexpr int HSTR = 68;   // padded f16 row stride for h   (bank-conflict pad)
constexpr int ZSTR = 196;  // padded f32 row stride for hz / xz_dec

// -------- LDS layout (bytes, all 16B aligned) --------
constexpr size_t OFF_SWE  = 0;        // enc_Wh swizzled f16     24576
constexpr size_t OFF_SWDH = 24576;    // dec_Wh swizzled f16     24576
constexpr size_t OFF_SWDI = 49152;    // dec_Wi swizzled f16     24576
constexpr size_t OFF_H    = 73728;    // h state f16 [64][68]     8704
constexpr size_t OFF_HZ   = 82432;    // hz f32 [64][196]        50176
constexpr size_t OFF_XZD  = 132608;   // xz_dec f32 [64][196]    50176
constexpr size_t OFF_XC   = 182784;   // x per-cell f32 [64][12]  3072
constexpr size_t OFF_WIE  = 185856;   // enc_Wi f32 [192]          768
constexpr size_t OFF_BIE  = 186624;   // enc_bi                    768
constexpr size_t OFF_BHE  = 187392;   // enc_bh                    768
constexpr size_t OFF_BID  = 188160;   // dec_bi                    768
constexpr size_t OFF_BHD  = 188928;   // dec_bh                    768
constexpr size_t OFF_WD   = 189696;   // Wd f32 [64]               256
constexpr size_t OFF_BD   = 189952;   // bd                         16
constexpr size_t SMEM_BYTES = 189968;

typedef __attribute__((ext_vector_type(16))) _Float16 v16h;
typedef __attribute__((ext_vector_type(8)))  float    v8f;

#define DEV __device__ __forceinline__

DEV float sigmoidf_(float x) { return 1.0f / (1.0f + __expf(-x)); }

union AFrag { v16h v; unsigned u[8]; };
union PackH { _Float16 f[2]; unsigned u; };

// Pre-swizzle a [64 x 192] f32 global weight matrix into the WMMA B-operand
// lane layout, f16, in LDS.  24 (ntile,ktile) pairs, 3 per wave.
// B tile (32x16), this code's contiguous k order (same as A): lanes 0-15 cover K=0..15 (VGPR v -> K=2v,2v+1),
// lanes 16-31 cover K=16..31; N = lane & 15.
DEV void load_swizzled_w(const float* __restrict__ Wg, unsigned* swiz,
                         int wave, int lane) {
  #pragma unroll
  for (int pi = 0; pi < 3; ++pi) {
    const int p  = wave * 3 + pi;          // p = ntile*2 + ktile, 0..23
    const int nt = p >> 1, kt = p & 1;
    const int col = nt * 16 + (lane & 15);
    const int kb  = kt * 32 + ((lane >> 4) << 4);
    #pragma unroll
    for (int v = 0; v < 8; ++v) {
      const int k = kb + 2 * v;
      PackH pk;
      pk.f[0] = (_Float16)Wg[(size_t)k       * K3D + col];
      pk.f[1] = (_Float16)Wg[(size_t)(k + 1) * K3D + col];
      swiz[(p * 32 + lane) * 8 + v] = pk.u;
    }
  }
}

// One recurrent GEMM: outz[64][192] = h[64][64](f16) @ W[64][192](swizzled f16)
// 8 waves: wave owns mtile = wave&3, ntiles [(wave>>2)*6, +6).
DEV void gemm_h_w(const unsigned* __restrict__ swiz,
                  const _Float16* __restrict__ hbuf,
                  float* __restrict__ outz, int wave, int lane) {
  const int mt     = wave & 3;
  const int ntBase = (wave >> 2) * 6;
  const int lh     = lane >> 4;             // lane half
  const int row    = mt * 16 + (lane & 15);

  // A operand: 16x32 f16 tile; lane holds row `row`, VGPR v holds K pair
  // kk = 2*v + lh*16  (pairs are contiguous -> one b32 load)
  AFrag a0, a1;
  #pragma unroll
  for (int v = 0; v < 8; ++v) {
    const int kk = 2 * v + lh * 16;
    a0.u[v] = *(const unsigned*)(hbuf + row * HSTR + kk);
    a1.u[v] = *(const unsigned*)(hbuf + row * HSTR + 32 + kk);
  }

  #pragma unroll
  for (int nt = 0; nt < 6; ++nt) {
    v8f acc = {};
    const int p0 = (ntBase + nt) * 2;
    const v16h b0 = *(const v16h*)(swiz + ((p0    ) * 32 + lane) * 8);
    const v16h b1 = *(const v16h*)(swiz + ((p0 + 1) * 32 + lane) * 8);
    acc = __builtin_amdgcn_wmma_f32_16x16x32_f16(false, a0.v, false, b0,
                                                 (short)0, acc, false, false);
    acc = __builtin_amdgcn_wmma_f32_16x16x32_f16(false, a1.v, false, b1,
                                                 (short)0, acc, false, false);
    // C/D layout: VGPR r -> M = r + 8*lh ; N = lane&15
    const int col   = (ntBase + nt) * 16 + (lane & 15);
    const int rbase = mt * 16 + 8 * lh;
    #pragma unroll
    for (int r = 0; r < 8; ++r)
      outz[(rbase + r) * ZSTR + col] = acc[r];
  }
}

__global__ void __launch_bounds__(256)
gru_seq2seq_cell_kernel(const float* __restrict__ Xg,
                        const float* __restrict__ encWi,
                        const float* __restrict__ encWh,
                        const float* __restrict__ encBi,
                        const float* __restrict__ encBh,
                        const float* __restrict__ decWi,
                        const float* __restrict__ decWh,
                        const float* __restrict__ decBi,
                        const float* __restrict__ decBh,
                        const float* __restrict__ Wdg,
                        const float* __restrict__ bdg,
                        float* __restrict__ out) {
  extern __shared__ __align__(16) char smem[];
  const int n    = blockIdx.x;                 // cell id
  const int tid  = threadIdx.x;
  const int wave = tid >> 5;
  const int lane = tid & 31;

  unsigned*  swE  = (unsigned*)(smem + OFF_SWE);
  unsigned*  swDh = (unsigned*)(smem + OFF_SWDH);
  unsigned*  swDi = (unsigned*)(smem + OFF_SWDI);
  _Float16*  hbuf = (_Float16*)(smem + OFF_H);
  float*     hz   = (float*)(smem + OFF_HZ);
  float*     xzd  = (float*)(smem + OFF_XZD);
  float*     xc   = (float*)(smem + OFF_XC);
  float*     WiE  = (float*)(smem + OFF_WIE);
  float*     biE  = (float*)(smem + OFF_BIE);
  float*     bhE  = (float*)(smem + OFF_BHE);
  float*     biD  = (float*)(smem + OFF_BID);
  float*     bhD  = (float*)(smem + OFF_BHD);
  float*     WdL  = (float*)(smem + OFF_WD);
  float*     bdL  = (float*)(smem + OFF_BD);

  // ---- load this cell's weights into LDS (stay resident for all 24 steps) ----
  load_swizzled_w(encWh + (size_t)n * Dd * K3D, swE,  wave, lane);
  load_swizzled_w(decWh + (size_t)n * Dd * K3D, swDh, wave, lane);
  load_swizzled_w(decWi + (size_t)n * Dd * K3D, swDi, wave, lane);
  if (tid < K3D) {
    WiE[tid] = encWi[(size_t)n * K3D + tid];
    biE[tid] = encBi[(size_t)n * K3D + tid];
    bhE[tid] = encBh[(size_t)n * K3D + tid];
    biD[tid] = decBi[(size_t)n * K3D + tid];
    bhD[tid] = decBh[(size_t)n * K3D + tid];
  }
  if (tid < Dd) WdL[tid] = Wdg[(size_t)n * Dd + tid];
  if (tid == 0) bdL[0] = bdg[n];
  for (int i = tid; i < Bb * Pp; i += 256) {           // X[b][t][cell n]
    const int b = i / Pp, t = i % Pp;
    xc[i] = Xg[(size_t)(b * Pp + t) * Nn + n];
  }
  for (int i = tid; i < Bb * HSTR; i += 256) hbuf[i] = (_Float16)0.0f;

  float hreg[16];
  #pragma unroll
  for (int i = 0; i < 16; ++i) hreg[i] = 0.0f;
  const int bRow  = tid >> 2;          // each thread owns 16 d's of one batch row
  const int dBase = (tid & 3) * 16;

  __syncthreads();

  // ---------------- encoder: P steps ----------------
  for (int t = 0; t < Pp; ++t) {
    gemm_h_w(swE, hbuf, hz, wave, lane);
    __syncthreads();
    const float xv = xc[bRow * Pp + t];
    #pragma unroll
    for (int i = 0; i < 16; ++i) {
      const int d = dBase + i;
      const float hz_z = hz[bRow * ZSTR + d]          + bhE[d];
      const float hz_r = hz[bRow * ZSTR + Dd + d]     + bhE[Dd + d];
      const float hz_g = hz[bRow * ZSTR + 2 * Dd + d] + bhE[2 * Dd + d];
      const float z = sigmoidf_(xv * WiE[d]          + biE[d]          + hz_z);
      const float r = sigmoidf_(xv * WiE[Dd + d]     + biE[Dd + d]     + hz_r);
      const float g = tanhf    (xv * WiE[2 * Dd + d] + biE[2 * Dd + d] + r * hz_g);
      hreg[i] = z * hreg[i] + (1.0f - z) * g;
      hbuf[bRow * HSTR + d] = (_Float16)hreg[i];
    }
    __syncthreads();
  }

  // ---------------- xz_dec = h_enc @ dec_Wi (bias added per step) ----------------
  gemm_h_w(swDi, hbuf, xzd, wave, lane);
  __syncthreads();

  // reset decoder state to zero
  #pragma unroll
  for (int i = 0; i < 16; ++i) hreg[i] = 0.0f;
  for (int i = tid; i < Bb * HSTR; i += 256) hbuf[i] = (_Float16)0.0f;
  __syncthreads();

  // ---------------- decoder: Q steps + fused Dense(1) ----------------
  for (int q = 0; q < Qq; ++q) {
    gemm_h_w(swDh, hbuf, hz, wave, lane);
    __syncthreads();
    float part = 0.0f;
    #pragma unroll
    for (int i = 0; i < 16; ++i) {
      const int d = dBase + i;
      const float hz_z = hz[bRow * ZSTR + d]          + bhD[d];
      const float hz_r = hz[bRow * ZSTR + Dd + d]     + bhD[Dd + d];
      const float hz_g = hz[bRow * ZSTR + 2 * Dd + d] + bhD[2 * Dd + d];
      const float z = sigmoidf_(xzd[bRow * ZSTR + d]          + biD[d]          + hz_z);
      const float r = sigmoidf_(xzd[bRow * ZSTR + Dd + d]     + biD[Dd + d]     + hz_r);
      const float g = tanhf    (xzd[bRow * ZSTR + 2 * Dd + d] + biD[2 * Dd + d] + r * hz_g);
      hreg[i] = z * hreg[i] + (1.0f - z) * g;
      hbuf[bRow * HSTR + d] = (_Float16)hreg[i];
      part += hreg[i] * WdL[d];
    }
    // reduce the Dense(1) dot over the 4 threads sharing batch row bRow
    part += __shfl_xor(part, 1, 32);
    part += __shfl_xor(part, 2, 32);
    if ((tid & 3) == 0)
      out[(size_t)(bRow * Qq + q) * Nn + n] = part + bdL[0];
    __syncthreads();
  }
}

extern "C" void kernel_launch(void* const* d_in, const int* in_sizes, int n_in,
                              void* d_out, int out_size, void* d_ws, size_t ws_size,
                              hipStream_t stream) {
  (void)in_sizes; (void)n_in; (void)out_size; (void)d_ws; (void)ws_size;
  const float* X     = (const float*)d_in[0];
  // d_in[1] = TE — unused by the reference computation
  const float* encWi = (const float*)d_in[2];
  const float* encWh = (const float*)d_in[3];
  const float* encBi = (const float*)d_in[4];
  const float* encBh = (const float*)d_in[5];
  const float* decWi = (const float*)d_in[6];
  const float* decWh = (const float*)d_in[7];
  const float* decBi = (const float*)d_in[8];
  const float* decBh = (const float*)d_in[9];
  const float* Wd    = (const float*)d_in[10];
  const float* bd    = (const float*)d_in[11];
  float* out = (float*)d_out;

  dim3 grid(Nn), block(256);
  gru_seq2seq_cell_kernel<<<grid, block, SMEM_BYTES, stream>>>(
      X, encWi, encWh, encBi, encBh, decWi, decWh, decBi, decBh, Wd, bd, out);
}
